// UpSample_24739011624967
// MI455X (gfx1250) — matrix operand order repaired and checked by reading, not compiled, not run on hardware
//
#include <hip/hip_runtime.h>

typedef __attribute__((ext_vector_type(16))) __bf16 v16bf;
typedef __attribute__((ext_vector_type(8)))  __bf16 v8bf;
typedef __attribute__((ext_vector_type(8)))  float  v8f;

// ---------------- helpers ----------------
__device__ __forceinline__ __bf16 f2bf(float f) {
  union { float f; unsigned u; } in; in.f = f;
  unsigned r = in.u + 0x7FFFu + ((in.u >> 16) & 1u);   // round-to-nearest-even
  unsigned short h = (unsigned short)(r >> 16);
  union { unsigned short s; __bf16 b; } out; out.s = h;
  return out.b;
}

// Problem constants (from reference setup_inputs)
#define BS   16
#define N1   4096
#define N2   1024
#define C1   128
#define C2   256
#define CIN  384
#define CH   256
#define MROWS (BS * N1)   // 65536

// ---------------------------------------------------------------------------
// Kernel 1: pack f32 weights [K,256] into per-lane WMMA bf16 B-fragment layout
// Wp[((kt*16 + nt)*32 + lane)*16 + e]  with:
//   n = nt*16 + (lane&15);  k = kt*32 + (lane>=16 ? 8 : 0) + (e&7) + (e>=8 ? 16 : 0)
// ---------------------------------------------------------------------------
__global__ __launch_bounds__(256) void pack_w_kernel(const float* __restrict__ W,
                                                     __bf16* __restrict__ Wp, int K) {
  int tid = blockIdx.x * 256 + threadIdx.x;
  if (tid >= K * 256) return;
  int e    = tid & 15;
  int lane = (tid >> 4) & 31;
  int frag = tid >> 9;
  int nt   = frag & 15;     // N/16 == 16
  int kt   = frag >> 4;
  int n = nt * 16 + (lane & 15);
  int k = kt * 32 + ((lane >= 16) ? 8 : 0) + (e & 7) + ((e >= 8) ? 16 : 0);
  Wp[tid] = f2bf(W[(size_t)k * 256 + n]);
}

// ---------------------------------------------------------------------------
// Kernel 2: fused 3-NN + inverse-distance interpolation + concat -> bf16 A1
// One wave (32 lanes) per query point. A1 is [65536 x 384] row-major bf16.
// ---------------------------------------------------------------------------
__global__ __launch_bounds__(256) void knn_build_a_kernel(
    const float* __restrict__ xyz1, const float* __restrict__ xyz2,
    const float* __restrict__ fea1, const float* __restrict__ fea2,
    __bf16* __restrict__ A1) {
  const int lane  = threadIdx.x & 31;
  const int wave  = threadIdx.x >> 5;
  const int point = blockIdx.x * 8 + wave;   // 0..65535
  const int b = point >> 12;                 // /4096

  const float px = xyz1[point * 3 + 0];
  const float py = xyz1[point * 3 + 1];
  const float pz = xyz1[point * 3 + 2];

  float d0 = 3.4e38f, d1 = 3.4e38f, d2 = 3.4e38f;
  int   i0 = 0, i1 = 0, i2 = 0;
  auto ins = [&](float d, int j) {
    if (d < d2) {
      if (d < d1) {
        if (d < d0) { d2 = d1; i2 = i1; d1 = d0; i1 = i0; d0 = d; i0 = j; }
        else        { d2 = d1; i2 = i1; d1 = d;  i1 = j; }
      } else        { d2 = d;  i2 = j; }
    }
  };

  const float* q = xyz2 + (size_t)b * N2 * 3;
  #pragma unroll 4
  for (int t = 0; t < N2 / 32; ++t) {
    int j = t * 32 + lane;
    float dx = q[j * 3 + 0] - px;
    float dy = q[j * 3 + 1] - py;
    float dz = q[j * 3 + 2] - pz;
    ins(dx * dx + dy * dy + dz * dz, j);
  }
  // wave32 butterfly merge of per-lane top-3 -> all lanes hold global top-3
  for (int off = 16; off; off >>= 1) {
    float e0 = __shfl_xor(d0, off, 32);
    float e1 = __shfl_xor(d1, off, 32);
    float e2 = __shfl_xor(d2, off, 32);
    int   j0 = __shfl_xor(i0, off, 32);
    int   j1 = __shfl_xor(i1, off, 32);
    int   j2 = __shfl_xor(i2, off, 32);
    ins(e0, j0); ins(e1, j1); ins(e2, j2);
  }

  float r0 = 1.0f / (d0 + 1e-8f);
  float r1 = 1.0f / (d1 + 1e-8f);
  float r2 = 1.0f / (d2 + 1e-8f);
  float s  = r0 + r1 + r2;
  float w0 = r0 / s, w1 = r1 / s, w2 = r2 / s;

  __bf16* arow = A1 + (size_t)point * CIN;
  const float* f1 = fea1 + (size_t)point * C1;
  #pragma unroll
  for (int c = lane; c < C1; c += 32) arow[c] = f2bf(f1[c]);

  const float* fb = fea2 + (size_t)b * N2 * C2;
  const float* g0 = fb + (size_t)i0 * C2;
  const float* g1 = fb + (size_t)i1 * C2;
  const float* g2 = fb + (size_t)i2 * C2;
  #pragma unroll
  for (int c = lane; c < C2; c += 32)
    arow[C1 + c] = f2bf(w0 * g0[c] + w1 * g1[c] + w2 * g2[c]);
}

// ---------------------------------------------------------------------------
// Kernel 3: fully fused MLP:  out = relu(relu(A x W1 + b1) x W2 + b2)
// Block = 128 threads = 4 waves; each wave owns 16 rows x full N=256.
// The hidden activation tile lives in LDS (8 KB per wave) and never touches
// HBM. W fragments are staged per-K-step into a shared 16 KB LDS buffer.
// B fragments are register-double-buffered so ds_load latency hides under
// the previous v_wmma.
// ---------------------------------------------------------------------------
__global__ __launch_bounds__(128) void fused_mlp_kernel(
    const __bf16* __restrict__ A, const __bf16* __restrict__ W1p,
    const float* __restrict__ b1, const __bf16* __restrict__ W2p,
    const float* __restrict__ b2, float* __restrict__ out) {
  __shared__ v16bf  sW[16 * 32];        // 16 KB: B fragments for one K-step
  __shared__ __bf16 sH[4 * 16 * 256];   // 32 KB: hidden tile, 8 KB per wave

  const int lane = threadIdx.x & 31;
  const int wave = threadIdx.x >> 5;
  const int row0 = blockIdx.x * 64 + wave * 16;
  const bool hi  = lane >= 16;
  const int lrow = lane & 15;
  const int mrow = row0 + lrow;         // A-fragment row for this lane

  v8f acc[16];

  // ================= GEMM1: h = relu(A[.,384] * W1 + b1) =================
  #pragma unroll
  for (int nt = 0; nt < 16; ++nt) {
    float bv = b1[nt * 16 + lrow];
    #pragma unroll
    for (int r = 0; r < 8; ++r) acc[nt][r] = bv;
  }
  for (int kt = 0; kt < CIN / 32; ++kt) {
    // issue A-fragment global loads before the staging barrier
    const __bf16* ap = A + (size_t)mrow * CIN + kt * 32 + (hi ? 8 : 0);
    v8bf c0 = *(const v8bf*)ap;
    v8bf c1 = *(const v8bf*)(ap + 16);
    __syncthreads();
    {  // stage 16 KB of packed W1 fragments: 1024 x b128 over 128 threads
      const uint4* src = (const uint4*)(W1p + (size_t)kt * (16 * 32 * 16));
      uint4* dst = (uint4*)sW;
      #pragma unroll
      for (int j = 0; j < 8; ++j)
        dst[threadIdx.x + j * 128] = src[threadIdx.x + j * 128];
    }
    __syncthreads();
    v16bf afrag = __builtin_shufflevector(c0, c1, 0, 1, 2, 3, 4, 5, 6, 7,
                                          8, 9, 10, 11, 12, 13, 14, 15);
    v16bf bcur = sW[lane];
    #pragma unroll
    for (int nt = 0; nt < 16; ++nt) {
      v16bf bnxt = bcur;
      if (nt < 15) bnxt = sW[(nt + 1) * 32 + lane];   // prefetch before wmma
      acc[nt] = __builtin_amdgcn_wmma_f32_16x16x32_bf16(
          false, afrag, false, bcur, (short)0, acc[nt], false, false);
      bcur = bnxt;
    }
  }

  // ============ hidden tile -> LDS (ReLU + bf16), wave-local ============
  __bf16* hrow = sH + wave * (16 * 256);
  #pragma unroll
  for (int nt = 0; nt < 16; ++nt) {
    int col = nt * 16 + lrow;
    #pragma unroll
    for (int r = 0; r < 8; ++r) {
      float v = acc[nt][r];
      v = v > 0.0f ? v : 0.0f;
      int rl = r + (hi ? 8 : 0);
      hrow[rl * 256 + col] = f2bf(v);
    }
  }
  // same-wave LDS ops are in-order; compiler inserts s_wait_dscnt before reads

  // ================= GEMM2: out = relu(h * W2 + b2) =================
  #pragma unroll
  for (int nt = 0; nt < 16; ++nt) {
    float bv = b2[nt * 16 + lrow];
    #pragma unroll
    for (int r = 0; r < 8; ++r) acc[nt][r] = bv;
  }
  for (int kt = 0; kt < CH / 32; ++kt) {
    __syncthreads();   // also guarantees GEMM1 sW readers are done
    {
      const uint4* src = (const uint4*)(W2p + (size_t)kt * (16 * 32 * 16));
      uint4* dst = (uint4*)sW;
      #pragma unroll
      for (int j = 0; j < 8; ++j)
        dst[threadIdx.x + j * 128] = src[threadIdx.x + j * 128];
    }
    __syncthreads();
    // A fragment straight from LDS hidden tile
    const __bf16* hp = hrow + lrow * 256 + kt * 32 + (hi ? 8 : 0);
    v8bf c0 = *(const v8bf*)hp;
    v8bf c1 = *(const v8bf*)(hp + 16);
    v16bf afrag = __builtin_shufflevector(c0, c1, 0, 1, 2, 3, 4, 5, 6, 7,
                                          8, 9, 10, 11, 12, 13, 14, 15);
    v16bf bcur = sW[lane];
    #pragma unroll
    for (int nt = 0; nt < 16; ++nt) {
      v16bf bnxt = bcur;
      if (nt < 15) bnxt = sW[(nt + 1) * 32 + lane];
      acc[nt] = __builtin_amdgcn_wmma_f32_16x16x32_bf16(
          false, afrag, false, bcur, (short)0, acc[nt], false, false);
      bcur = bnxt;
    }
  }

  // ================= epilogue: ReLU + f32 stores =================
  #pragma unroll
  for (int nt = 0; nt < 16; ++nt) {
    int col = nt * 16 + lrow;
    #pragma unroll
    for (int r = 0; r < 8; ++r) {
      int row = row0 + r + (hi ? 8 : 0);
      float v = acc[nt][r];
      out[(size_t)row * 256 + col] = v > 0.0f ? v : 0.0f;
    }
  }
}

// ---------------------------------------------------------------------------
extern "C" void kernel_launch(void* const* d_in, const int* in_sizes, int n_in,
                              void* d_out, int out_size, void* d_ws, size_t ws_size,
                              hipStream_t stream) {
  (void)in_sizes; (void)n_in; (void)out_size; (void)ws_size;
  const float* xyz1 = (const float*)d_in[0];
  const float* xyz2 = (const float*)d_in[1];
  const float* fea1 = (const float*)d_in[2];
  const float* fea2 = (const float*)d_in[3];
  const float* W1   = (const float*)d_in[4];
  const float* b1   = (const float*)d_in[5];
  const float* W2   = (const float*)d_in[6];
  const float* b2   = (const float*)d_in[7];

  // workspace layout (bytes)
  char* ws = (char*)d_ws;
  const size_t szA1 = (size_t)MROWS * CIN * 2;   // 50,331,648
  const size_t szW1 = (size_t)CIN * 256 * 2;     //    196,608
  __bf16* A1  = (__bf16*)(ws);
  __bf16* W1p = (__bf16*)(ws + szA1);
  __bf16* W2p = (__bf16*)(ws + szA1 + szW1);

  pack_w_kernel<<<(CIN * 256) / 256, 256, 0, stream>>>(W1, W1p, CIN);
  pack_w_kernel<<<(CH * 256) / 256, 256, 0, stream>>>(W2, W2p, CH);
  knn_build_a_kernel<<<MROWS / 8, 256, 0, stream>>>(xyz1, xyz2, fea1, fea2, A1);
  fused_mlp_kernel<<<MROWS / 64, 128, 0, stream>>>(A1, W1p, b1, W2p, b2,
                                                   (float*)d_out);
}
